// MultiHeadAttention_21964462752506
// MI455X (gfx1250) — compile-verified, matrix operand order checked by reading, not run on hardware
//
#include <hip/hip_runtime.h>

// ---------------------------------------------------------------------------
// MI455X (gfx1250) MHA forward: bf16 WMMA (16x16x32) with fp32 accumulation.
// wave32 everywhere.  Layouts per CDNA5 ISA 7.12.2:
//   A (16xK, 16-bit):  lane holds row M=lane%16; element e -> K = (e/8)*16 + (lane/16)*8 + (e%8)
//   B (Kx16, 16-bit):  lane holds col N=lane%16; element e -> K = same formula
//   C/D (16x16 f32):   VGPR r, lane -> M = r + (lane/16)*8, N = lane%16
// So both A and B fragments load as two contiguous 16-byte chunks from a
// row-major [outer x K] bf16 matrix -> we pre-transpose weights to W^T.
// V-tile staging in attention uses GLOBAL_LOAD_ASYNC_TO_LDS_B128 (ASYNCcnt).
// ---------------------------------------------------------------------------

typedef __bf16 bf16_t;
typedef __attribute__((ext_vector_type(16))) __bf16 v16bf;
typedef __attribute__((ext_vector_type(8)))  __bf16 v8bf;
typedef __attribute__((ext_vector_type(8)))  float  v8f;

#define D_MODEL 1024
#define NHEAD   16
#define HDIM    64
#define BATCH   2
#define SEQ     2048
#define MTOT    (BATCH * SEQ)   // 4096 rows

__device__ __forceinline__ v16bf frag_combine(v8bf lo, v8bf hi) {
  v16bf r;
#pragma unroll
  for (int i = 0; i < 8; ++i) { r[i] = lo[i]; r[i + 8] = hi[i]; }
  return r;
}

// Load an A/B-style 16x32 bf16 fragment from row-major [rows x stride].
// outer index = row (caller passes base_row + lane%16), K window = [kbase, kbase+32).
__device__ __forceinline__ v16bf load_frag(const bf16_t* p, int row, int stride,
                                           int kbase, int lane) {
  const bf16_t* rp = p + (size_t)row * stride + kbase + ((lane >> 4) << 3);
  v8bf lo = *reinterpret_cast<const v8bf*>(rp);
  v8bf hi = *reinterpret_cast<const v8bf*>(rp + 16);
  return frag_combine(lo, hi);
}

__device__ __forceinline__ v8f wmma_bf16(v16bf a, v16bf b, v8f c) {
  return __builtin_amdgcn_wmma_f32_16x16x32_bf16(false, a, false, b,
                                                 (short)0, c, false, false);
}

// ---------------------------------------------------------------------------
// Prep: fp32 -> bf16 conversion of x; transpose+convert of weights.
// ---------------------------------------------------------------------------
__global__ void convert_x_kernel(const float* __restrict__ x,
                                 bf16_t* __restrict__ xb, int n) {
  int i = blockIdx.x * blockDim.x + threadIdx.x;
  if (i < n) xb[i] = (bf16_t)x[i];
}

__global__ void transpose_w_kernel(const float* __restrict__ W,
                                   bf16_t* __restrict__ WT) {
  __shared__ float tile[32][33];
  int tx = threadIdx.x, ty = threadIdx.y;      // 32 x 8
  int n0 = blockIdx.x * 32, k0 = blockIdx.y * 32;
#pragma unroll
  for (int j = 0; j < 32; j += 8)
    tile[ty + j][tx] = W[(size_t)(k0 + ty + j) * D_MODEL + n0 + tx];
  __syncthreads();
#pragma unroll
  for (int j = 0; j < 32; j += 8)
    WT[(size_t)(n0 + ty + j) * D_MODEL + k0 + tx] = (bf16_t)tile[tx][ty + j];
}

// ---------------------------------------------------------------------------
// GEMM:  C[4096x1024] = A_bf16[4096x1024] @ (BT_bf16[1024x1024])^T + bias
// 256 threads = 8 waves; wave tile 32x64 (2x4 WMMA accumulators);
// block tile 64(M) x 256(N); K loop step 32.
// Per K-step: 6 fragment loads (12 x b128) feed 8 WMMAs.
// Epilogue optionally scatters into [B,H,S,64] (bf16 and/or fp32 `present`)
// and/or writes row-major fp32 (final projection).
// ---------------------------------------------------------------------------
__global__ __launch_bounds__(256) void gemm_kernel(
    const bf16_t* __restrict__ A, const bf16_t* __restrict__ BT,
    const float* __restrict__ bias,
    bf16_t* __restrict__ out_bhsd_bf,   // q/k/v bf16 in [B,H,S,64] (or null)
    float*  __restrict__ out_bhsd_f,    // present k/v fp32 in [B,H,S,64] (or null)
    float*  __restrict__ out_rm_f)      // row-major fp32 [4096x1024] (or null)
{
  const int wid  = threadIdx.x >> 5;
  const int lane = threadIdx.x & 31;
  const int half = lane >> 4;
  const int m0 = blockIdx.y * 64  + (wid >> 2) * 32;
  const int n0 = blockIdx.x * 256 + (wid & 3) * 64;

  v8f acc[2][4] = {};
  for (int kk = 0; kk < D_MODEL; kk += 32) {
    v16bf a0 = load_frag(A, m0      + (lane & 15), D_MODEL, kk, lane);
    v16bf a1 = load_frag(A, m0 + 16 + (lane & 15), D_MODEL, kk, lane);
#pragma unroll
    for (int ni = 0; ni < 4; ++ni) {
      v16bf b = load_frag(BT, n0 + ni * 16 + (lane & 15), D_MODEL, kk, lane);
      acc[0][ni] = wmma_bf16(a0, b, acc[0][ni]);
      acc[1][ni] = wmma_bf16(a1, b, acc[1][ni]);
    }
  }

#pragma unroll
  for (int mi = 0; mi < 2; ++mi)
#pragma unroll
    for (int ni = 0; ni < 4; ++ni)
#pragma unroll
      for (int r = 0; r < 8; ++r) {
        int m = m0 + mi * 16 + r + half * 8;
        int n = n0 + ni * 16 + (lane & 15);
        float val = acc[mi][ni][r] + bias[n];
        if (out_rm_f) out_rm_f[(size_t)m * D_MODEL + n] = val;
        if (out_bhsd_bf || out_bhsd_f) {
          int b = m >> 11, s = m & (SEQ - 1);
          int h = n >> 6,  dd = n & (HDIM - 1);
          size_t idx = (((size_t)(b * NHEAD + h)) * SEQ + s) * HDIM + dd;
          if (out_bhsd_bf) out_bhsd_bf[idx] = (bf16_t)val;
          if (out_bhsd_f)  out_bhsd_f[idx]  = val;
        }
      }
}

// ---------------------------------------------------------------------------
// Causal flash attention. 128 threads = 4 waves. Wave = 16 query rows, d=64.
// Per 32-key tile: 4 WMMA for scores, online softmax (shfl_xor row reduce),
// 4 WMMA for P@V.  V tile staged per-wave straight into LDS via
// GLOBAL_LOAD_ASYNC_TO_LDS_B128 (ASYNCcnt), fenced with s_wait_asynccnt just
// before the B-fragment gather so the copy overlaps the score WMMAs/softmax.
// No cross-wave barriers: per-wave trip counts diverge on the causal mask.
// ---------------------------------------------------------------------------
__global__ __launch_bounds__(128) void attn_kernel(
    const bf16_t* __restrict__ qb, const bf16_t* __restrict__ kb,
    const bf16_t* __restrict__ vb, bf16_t* __restrict__ attn_out)
{
  __shared__ __align__(16) bf16_t Vt[4][32][72];  // per-wave V tile (padded rows)
  __shared__ __align__(16) bf16_t Pt[4][16][32];  // per-wave P tile

  const int wid  = threadIdx.x >> 5;
  const int lane = threadIdx.x & 31;
  const int half = lane >> 4;
  const int bh = blockIdx.y;                 // b*NHEAD + h
  const int m0 = blockIdx.x * 64 + wid * 16; // query row base within S

  const bf16_t* qrow = qb + (size_t)bh * SEQ * HDIM;
  const bf16_t* krow = kb + (size_t)bh * SEQ * HDIM;
  const bf16_t* vrow = vb + (size_t)bh * SEQ * HDIM;

  v16bf Aq0 = load_frag(qrow, m0 + (lane & 15), HDIM, 0,  lane);
  v16bf Aq1 = load_frag(qrow, m0 + (lane & 15), HDIM, 32, lane);

  v8f O[4] = {};
  float mrow[8], lrow[8];
#pragma unroll
  for (int r = 0; r < 8; ++r) { mrow[r] = -1.0e30f; lrow[r] = 0.0f; }

  const float scale = 0.03125f;  // 1/sqrt(D_MODEL)
  const int kend = m0 + 15;

  for (int kt = 0; kt <= kend; kt += 32) {
    // ---- async-stage V tile (this wave only): lane -> key kt+lane ----
    // 8 x ASYNC_TO_LDS_B128: IOFFSET is added to BOTH the LDS and global
    // addresses (ISA 10.4.4), so offset:0..112 copies one 128B row.
    {
      const bf16_t* vp = vrow + (size_t)(kt + lane) * HDIM;
      unsigned lds_dst = (unsigned)(unsigned long long)(void*)&Vt[wid][lane][0];
      asm volatile(
          "global_load_async_to_lds_b128 %0, %1, off\n\t"
          "global_load_async_to_lds_b128 %0, %1, off offset:16\n\t"
          "global_load_async_to_lds_b128 %0, %1, off offset:32\n\t"
          "global_load_async_to_lds_b128 %0, %1, off offset:48\n\t"
          "global_load_async_to_lds_b128 %0, %1, off offset:64\n\t"
          "global_load_async_to_lds_b128 %0, %1, off offset:80\n\t"
          "global_load_async_to_lds_b128 %0, %1, off offset:96\n\t"
          "global_load_async_to_lds_b128 %0, %1, off offset:112"
          :: "v"(lds_dst), "v"((unsigned long long)vp) : "memory");
      __builtin_prefetch(vp + 32 * HDIM, 0, 3);   // next tile -> global_prefetch
    }

    // ---- scores: S = q(16x64) @ k^T(64x32), K-chained WMMA ----
    v8f S0 = {}, S1 = {};
    {
      v16bf bk0 = load_frag(krow, kt      + (lane & 15), HDIM, 0,  lane);
      v16bf bk1 = load_frag(krow, kt      + (lane & 15), HDIM, 32, lane);
      v16bf bk2 = load_frag(krow, kt + 16 + (lane & 15), HDIM, 0,  lane);
      v16bf bk3 = load_frag(krow, kt + 16 + (lane & 15), HDIM, 32, lane);
      S0 = wmma_bf16(Aq0, bk0, S0);
      S0 = wmma_bf16(Aq1, bk1, S0);
      S1 = wmma_bf16(Aq0, bk2, S1);
      S1 = wmma_bf16(Aq1, bk3, S1);
    }

    // ---- online softmax (fp32), rows live across 16-lane halves ----
#pragma unroll
    for (int r = 0; r < 8; ++r) {
      int row = m0 + r + half * 8;
      int c0 = kt + (lane & 15);
      int c1 = c0 + 16;
      float s0 = S0[r] * scale; if (c0 > row) s0 = -3.0e38f;
      float s1 = S1[r] * scale; if (c1 > row) s1 = -3.0e38f;
      float mx = fmaxf(s0, s1);
#pragma unroll
      for (int off = 1; off < 16; off <<= 1)
        mx = fmaxf(mx, __shfl_xor(mx, off, 32));
      float mnew  = fmaxf(mrow[r], mx);
      float alpha = __expf(mrow[r] - mnew);
      float p0 = __expf(s0 - mnew);
      float p1 = __expf(s1 - mnew);
      float sum = p0 + p1;
#pragma unroll
      for (int off = 1; off < 16; off <<= 1)
        sum += __shfl_xor(sum, off, 32);
      lrow[r] = lrow[r] * alpha + sum;
      mrow[r] = mnew;
#pragma unroll
      for (int n = 0; n < 4; ++n) O[n][r] = O[n][r] * alpha;
      // C-layout -> row-major P tile in LDS (bf16) for A-fragment reload
      Pt[wid][r + half * 8][(lane & 15)]      = (bf16_t)p0;
      Pt[wid][r + half * 8][(lane & 15) + 16] = (bf16_t)p1;
    }

    // ---- fence the async V copy, then O += P(16x32) @ V(32x64) ----
    asm volatile("s_wait_asynccnt 0" ::: "memory");
    v16bf Pf = load_frag(&Pt[wid][0][0], lane & 15, 32, 0, lane);
#pragma unroll
    for (int n = 0; n < 4; ++n) {
      v16bf bv;
#pragma unroll
      for (int e = 0; e < 16; ++e) {
        int klocal = ((e >> 3) << 4) + (half << 3) + (e & 7);
        bv[e] = Vt[wid][klocal][n * 16 + (lane & 15)];
      }
      O[n] = wmma_bf16(Pf, bv, O[n]);
    }
  }

  // ---- normalize + write attn (bf16, row-major [B*S, D]) ----
  const int b = bh >> 4, h = bh & 15;
#pragma unroll
  for (int r = 0; r < 8; ++r) {
    float rl = 1.0f / lrow[r];
    int s = m0 + r + half * 8;
    size_t rowbase = ((size_t)(b * SEQ + s)) * D_MODEL + h * HDIM;
#pragma unroll
    for (int n = 0; n < 4; ++n)
      attn_out[rowbase + n * 16 + (lane & 15)] = (bf16_t)(O[n][r] * rl);
  }
}

// ---------------------------------------------------------------------------
// Host-side orchestration (all on `stream`, graph-capture safe).
// d_in order: x, Wq, bq, Wk, bk, Wv, bv, Wo, bo
// d_out: out [2,2048,1024] fp32, then present [2,2,16,2048,64] fp32
// ---------------------------------------------------------------------------
extern "C" void kernel_launch(void* const* d_in, const int* in_sizes, int n_in,
                              void* d_out, int out_size, void* d_ws, size_t ws_size,
                              hipStream_t stream) {
  const float* x  = (const float*)d_in[0];
  const float* Wq = (const float*)d_in[1];
  const float* bq = (const float*)d_in[2];
  const float* Wk = (const float*)d_in[3];
  const float* bk = (const float*)d_in[4];
  const float* Wv = (const float*)d_in[5];
  const float* bv = (const float*)d_in[6];
  const float* Wo = (const float*)d_in[7];
  const float* bo = (const float*)d_in[8];

  float* out = (float*)d_out;
  float* present_k = out + (size_t)MTOT * D_MODEL;          // +4M floats
  float* present_v = present_k + (size_t)MTOT * D_MODEL;    // +8M floats

  const size_t WSZ = (size_t)D_MODEL * D_MODEL;  // 1M elements
  const size_t XSZ = (size_t)MTOT * D_MODEL;     // 4M elements
  bf16_t* WqT   = (bf16_t*)d_ws;
  bf16_t* WkT   = WqT + WSZ;
  bf16_t* WvT   = WkT + WSZ;
  bf16_t* WoT   = WvT + WSZ;
  bf16_t* xb    = WoT + WSZ;
  bf16_t* qbf   = xb  + XSZ;
  bf16_t* kbf   = qbf + XSZ;
  bf16_t* vbf   = kbf + XSZ;
  bf16_t* attnb = vbf + XSZ;   // total ws: 4*1M + 5*4M bf16 = 48 MB

  dim3 tb(32, 8), tg(D_MODEL / 32, D_MODEL / 32);
  transpose_w_kernel<<<tg, tb, 0, stream>>>(Wq, WqT);
  transpose_w_kernel<<<tg, tb, 0, stream>>>(Wk, WkT);
  transpose_w_kernel<<<tg, tb, 0, stream>>>(Wv, WvT);
  transpose_w_kernel<<<tg, tb, 0, stream>>>(Wo, WoT);
  convert_x_kernel<<<(int)((XSZ + 255) / 256), 256, 0, stream>>>(x, xb, (int)XSZ);

  dim3 gg(D_MODEL / 256, MTOT / 64);
  gemm_kernel<<<gg, 256, 0, stream>>>(xb, WqT, bq, qbf, nullptr,  nullptr);
  gemm_kernel<<<gg, 256, 0, stream>>>(xb, WkT, bk, kbf, present_k, nullptr);
  gemm_kernel<<<gg, 256, 0, stream>>>(xb, WvT, bv, vbf, present_v, nullptr);

  attn_kernel<<<dim3(SEQ / 64, BATCH * NHEAD), 128, 0, stream>>>(qbf, kbf, vbf, attnb);

  gemm_kernel<<<gg, 256, 0, stream>>>(attnb, WoT, bo, nullptr, nullptr, out);
}
